// MultiHeadAttentionParrallel_41875931136749
// MI455X (gfx1250) — compile-verified, matrix operand order checked by reading, not run on hardware
//
#include <hip/hip_runtime.h>

// MI455X / gfx1250: wave32, WMMA 16x16x32 f16 -> f32 accumulate.
typedef __attribute__((ext_vector_type(16))) _Float16 v16h;
typedef __attribute__((ext_vector_type(8)))  _Float16 v8h;
typedef __attribute__((ext_vector_type(8)))  float    v8f;

#define WMMA_F16F32(a, b, c) \
  __builtin_amdgcn_wmma_f32_16x16x32_f16(false, (a), false, (b), (short)0, (c), false, false)

// Async global->LDS (ASYNCcnt-tracked, GVS mode: 64-bit SGPR base + 32-bit VGPR offset).
#define ASYNC_LD_B128(ldsaddr, voff, sbase)                       \
  asm volatile("global_load_async_to_lds_b128 %0, %1, %2"         \
               :: "v"(ldsaddr), "v"(voff), "s"(sbase) : "memory")
#define WAIT_ASYNC(n) asm volatile("s_wait_asynccnt " #n ::: "memory")

constexpr int Bb = 4;
constexpr int S  = 1024;
constexpr int E  = 1024;
constexpr int H  = 16;
constexpr int Dh = 64;            // head dim
constexpr int Mrows = Bb * S;     // 4096

// ---------------------------------------------------------------------------
// DPP ROW_XMASK xor-shuffle within 16-lane rows
// ---------------------------------------------------------------------------
template <int MASK>
__device__ __forceinline__ float dpp_xor16(float x) {
  int yi = __builtin_amdgcn_update_dpp(0, __builtin_bit_cast(int, x),
                                       0x160 | MASK, 0xF, 0xF, true);
  return __builtin_bit_cast(float, yi);
}
__device__ __forceinline__ float rowmax16(float v) {
  v = fmaxf(v, dpp_xor16<1>(v));
  v = fmaxf(v, dpp_xor16<2>(v));
  v = fmaxf(v, dpp_xor16<4>(v));
  v = fmaxf(v, dpp_xor16<8>(v));
  return v;
}
__device__ __forceinline__ float rowsum16(float v) {
  v += dpp_xor16<1>(v);
  v += dpp_xor16<2>(v);
  v += dpp_xor16<4>(v);
  v += dpp_xor16<8>(v);
  return v;
}

__device__ __forceinline__ unsigned lds_off(const void* p) {
  // Flat shared address: aperture in [63:32], LDS byte offset in [31:0] (ISA 10.2).
  return (unsigned)(uintptr_t)p;
}

// ---------------------------------------------------------------------------
// f32 -> f16 conversion
// ---------------------------------------------------------------------------
__global__ __launch_bounds__(256) void cvt_f32_to_f16(const float* __restrict__ src,
                                                      _Float16* __restrict__ dst, int n) {
  int i = blockIdx.x * 256 + threadIdx.x;
  if (i < n) dst[i] = (_Float16)src[i];
}

// ---------------------------------------------------------------------------
// Tiled NT GEMM: C[M,N] = A[M,K] * B[N,K]^T   (both f16 row-major, K-contiguous)
// Block tile 128x128, K step 32, 8 waves: wave tile 32x64 (2x4 WMMA frags).
// Double-buffered LDS fed by async global->LDS (ASYNCcnt pipelining).
// OUTMODE 0: plain f32 [M,N]                      (final output projection)
// OUTMODE 1: f16 head-split [B,H,S,D]             (Q/K projections)
// OUTMODE 2: f16 head-split transposed [B,H,D,S]  (V projection; packed stores)
// ---------------------------------------------------------------------------
template <int OUTMODE>
__global__ __launch_bounds__(256) void gemm_nt_f16(const _Float16* __restrict__ A,
                                                   const _Float16* __restrict__ Bm,
                                                   void* __restrict__ outp,
                                                   int M, int N, int K) {
  __shared__ _Float16 Al[2][128 * 32];
  __shared__ _Float16 Bl[2][128 * 32];

  const int tid   = threadIdx.x;
  const int w     = tid >> 5;
  const int lane  = tid & 31;
  const int lc    = lane & 15;
  const int hi    = lane >> 4;
  const int waveM = w & 3;   // 4 waves along M (32 rows each)
  const int waveN = w >> 2;  // 2 waves along N (64 cols each)
  const int blockN = blockIdx.x * 128;
  const int blockM = blockIdx.y * 128;

  v8f acc[2][4] = {};

  const int ldrow = tid >> 1;  // 0..127

  // Stage tile t (K cols k0..k0+31) into buffer t&1: 4 async b128 ops per thread.
  auto stage = [&](int t) {
    const int buf = t & 1;
    const int k0  = t * 32;
#pragma unroll
    for (int j = 0; j < 2; ++j) {
      const int kh = (tid & 1) * 16 + j * 8;  // half-offset within 32-wide K slab
      unsigned la = lds_off(&Al[buf][ldrow * 32 + kh]);
      unsigned lb = lds_off(&Bl[buf][ldrow * 32 + kh]);
      unsigned ga = (unsigned)(((size_t)(blockM + ldrow) * K + k0 + kh) * 2);
      unsigned gb = (unsigned)(((size_t)(blockN + ldrow) * K + k0 + kh) * 2);
      ASYNC_LD_B128(la, ga, A);
      ASYNC_LD_B128(lb, gb, Bm);
    }
  };

  const int T = K / 32;
  stage(0);
  for (int t = 0; t < T; ++t) {
    const int buf = t & 1;
    if (t + 1 < T) {
      stage(t + 1);
      WAIT_ASYNC(0x4);  // older 4 ops (stage t) complete; stage t+1 stays in flight
    } else {
      WAIT_ASYNC(0x0);
    }
    __syncthreads();

    v16h af[2], bf[4];
#pragma unroll
    for (int mt = 0; mt < 2; ++mt)
      af[mt] = *(const v16h*)&Al[buf][(waveM * 32 + mt * 16 + lc) * 32 + hi * 16];
#pragma unroll
    for (int nt = 0; nt < 4; ++nt)
      bf[nt] = *(const v16h*)&Bl[buf][(waveN * 64 + nt * 16 + lc) * 32 + hi * 16];

#pragma unroll
    for (int mt = 0; mt < 2; ++mt)
#pragma unroll
      for (int nt = 0; nt < 4; ++nt)
        acc[mt][nt] = WMMA_F16F32(af[mt], bf[nt], acc[mt][nt]);
    __syncthreads();
  }

  // epilogue: C-layout element (Mr, Nc): VGPR r, lanes0-15 -> Mr=r, lanes16-31 -> Mr=r+8
  if (OUTMODE == 2) {
    // V: [B,H,D,S]; per fragment the 8 accumulator rows are 8 consecutive s -> one b128 store
#pragma unroll
    for (int mt = 0; mt < 2; ++mt)
#pragma unroll
      for (int nt = 0; nt < 4; ++nt) {
        v8h p;
#pragma unroll
        for (int r = 0; r < 8; ++r) p[r] = (_Float16)acc[mt][nt][r];
        int gm = blockM + waveM * 32 + mt * 16 + 8 * hi;  // s base (8-aligned)
        int gn = blockN + waveN * 64 + nt * 16 + lc;
        int b = gm >> 10, s = gm & 1023;
        int h = gn >> 6,  d = gn & 63;
        *(v8h*)((_Float16*)outp + (((size_t)(b * H + h)) * Dh + d) * S + s) = p;
      }
  } else {
#pragma unroll
    for (int mt = 0; mt < 2; ++mt)
#pragma unroll
      for (int nt = 0; nt < 4; ++nt)
#pragma unroll
        for (int r = 0; r < 8; ++r) {
          int gm = blockM + waveM * 32 + mt * 16 + r + 8 * hi;
          int gn = blockN + waveN * 64 + nt * 16 + lc;
          float v = acc[mt][nt][r];
          if (OUTMODE == 0) {
            ((float*)outp)[(size_t)gm * N + gn] = v;
          } else {
            int b = gm >> 10, s = gm & 1023;
            int h = gn >> 6,  d = gn & 63;
            ((_Float16*)outp)[(((size_t)(b * H + h)) * S + s) * Dh + d] = (_Float16)v;
          }
        }
  }
}

// ---------------------------------------------------------------------------
// Flash attention: one block per (b, h, 128-query tile); 8 waves x 16 queries.
// K [B,H,S,D] and V^T [B,H,D,S] tiles double-buffered via async global->LDS.
// Online softmax (DPP row reductions), context in f32 WMMA accumulators.
// ---------------------------------------------------------------------------
__global__ __launch_bounds__(256) void flash_attn_f16(const _Float16* __restrict__ Qh,
                                                      const _Float16* __restrict__ Kh,
                                                      const _Float16* __restrict__ Vt,
                                                      _Float16* __restrict__ ctx) {
  __shared__ _Float16 Kl[2][32 * 64];   // [key][d]
  __shared__ _Float16 Vl[2][64 * 32];   // [d][key]  (already transposed in memory)
  __shared__ _Float16 Pst[8][16 * 32];  // per-wave P staging (C-layout -> A-layout)

  const int bid = blockIdx.x;
  const int qt = bid & 7;
  const int h  = (bid >> 3) & 15;
  const int b  = bid >> 7;
  const size_t headOff = ((size_t)(b * H + h)) * S * Dh;
  const _Float16* Kbase = Kh + headOff;
  const _Float16* Vbase = Vt + headOff;  // [D][S] within this head

  const int tid  = threadIdx.x;
  const int w    = tid >> 5;
  const int lane = tid & 31;
  const int lc   = lane & 15;
  const int hi   = lane >> 4;
  const int q0   = qt * 128 + w * 16;

  // Q fragments (A layout), row = q0 + lc, halves along d
  v16h qa[2];
#pragma unroll
  for (int kk = 0; kk < 2; ++kk)
    qa[kk] = *(const v16h*)(Qh + headOff + (size_t)(q0 + lc) * Dh + kk * 32 + hi * 16);

  float mrow[8], lrow[8];
#pragma unroll
  for (int r = 0; r < 8; ++r) { mrow[r] = -1e30f; lrow[r] = 0.f; }
  v8f acc[4] = {};

  const int krow = tid >> 3;        // 0..31: K-tile row this thread stages
  const int kd0  = (tid & 7) * 8;   // 8 halves along d
  const int vrow = tid >> 2;        // 0..63: V^T-tile row (d) this thread stages
  const int vk0  = (tid & 3) * 8;   // 8 halves along k

  // Stage key-tile t into buffer t&1: 2 async b128 ops per thread (K + V^T).
  auto stage = [&](int t) {
    const int buf = t & 1;
    const int k0  = t * 32;
    unsigned lk = lds_off(&Kl[buf][krow * 64 + kd0]);
    unsigned lv = lds_off(&Vl[buf][vrow * 32 + vk0]);
    unsigned gk = (unsigned)(((size_t)(k0 + krow) * Dh + kd0) * 2);
    unsigned gv = (unsigned)(((size_t)vrow * S + k0 + vk0) * 2);
    ASYNC_LD_B128(lk, gk, Kbase);
    ASYNC_LD_B128(lv, gv, Vbase);
  };

  const int T = S / 32;
  stage(0);
  for (int t = 0; t < T; ++t) {
    const int buf = t & 1;
    if (t + 1 < T) {
      stage(t + 1);
      WAIT_ASYNC(0x2);  // stage t landed; stage t+1 stays in flight
    } else {
      WAIT_ASYNC(0x0);
    }
    __syncthreads();

    // scores: S16x32 = Q(16x64) * K_tile(32x64)^T, two 16-key N tiles
    v8f sc0 = {}, sc1 = {};
#pragma unroll
    for (int kk = 0; kk < 2; ++kk) {
      v16h bf0 = *(const v16h*)&Kl[buf][(lc)      * 64 + kk * 32 + hi * 16];
      v16h bf1 = *(const v16h*)&Kl[buf][(16 + lc) * 64 + kk * 32 + hi * 16];
      sc0 = WMMA_F16F32(qa[kk], bf0, sc0);
      sc1 = WMMA_F16F32(qa[kk], bf1, sc1);
    }

    // scale 1/sqrt(D) + online softmax per row (rows r / r+8 per lane half)
#pragma unroll
    for (int r = 0; r < 8; ++r) { sc0[r] *= 0.125f; sc1[r] *= 0.125f; }
#pragma unroll
    for (int r = 0; r < 8; ++r) {
      float v   = rowmax16(fmaxf(sc0[r], sc1[r]));
      float mn  = fmaxf(mrow[r], v);
      float scl = __expf(mrow[r] - mn);
      float e0  = __expf(sc0[r] - mn);
      float e1  = __expf(sc1[r] - mn);
      float rs  = rowsum16(e0 + e1);
      lrow[r] = lrow[r] * scl + rs;
      mrow[r] = mn;
      acc[0][r] *= scl; acc[1][r] *= scl; acc[2][r] *= scl; acc[3][r] *= scl;
      int row = r + 8 * hi;
      Pst[w][row * 32 + lc]      = (_Float16)e0;
      Pst[w][row * 32 + 16 + lc] = (_Float16)e1;
    }

    // ctx += P(16x32) * V_tile(32x64): B frags straight from V^T rows
    v16h pa = *(const v16h*)&Pst[w][lc * 32 + hi * 16];
#pragma unroll
    for (int nt = 0; nt < 4; ++nt) {
      v16h bf = *(const v16h*)&Vl[buf][(nt * 16 + lc) * 32 + hi * 16];
      acc[nt] = WMMA_F16F32(pa, bf, acc[nt]);
    }
    __syncthreads();
  }

  // normalize + merged-head f16 store: ctx[b*S+s, h*64+d]
#pragma unroll
  for (int nt = 0; nt < 4; ++nt)
#pragma unroll
    for (int r = 0; r < 8; ++r) {
      int row = q0 + r + 8 * hi;
      int col = h * Dh + nt * 16 + lc;
      float o = acc[nt][r] / lrow[r];
      ctx[(size_t)(b * S + row) * E + col] = (_Float16)o;
    }
}

// ---------------------------------------------------------------------------
extern "C" void kernel_launch(void* const* d_in, const int* in_sizes, int n_in,
                              void* d_out, int out_size, void* d_ws, size_t ws_size,
                              hipStream_t stream) {
  (void)in_sizes; (void)n_in; (void)out_size; (void)ws_size;
  const float* q  = (const float*)d_in[0];
  const float* k  = (const float*)d_in[1];
  const float* v  = (const float*)d_in[2];
  const float* Wq = (const float*)d_in[3];
  const float* Wk = (const float*)d_in[4];
  const float* Wv = (const float*)d_in[5];
  const float* Wo = (const float*)d_in[6];
  float* out = (float*)d_out;

  // workspace layout (bytes), 56 MB total
  char* ws = (char*)d_ws;
  _Float16* Xq = (_Float16*)(ws + (size_t)0);          // 8MB  (reused as ctx later)
  _Float16* Xk = (_Float16*)(ws + ((size_t)8  << 20)); // 8MB
  _Float16* Xv = (_Float16*)(ws + ((size_t)16 << 20)); // 8MB
  _Float16* Wqh = (_Float16*)(ws + ((size_t)24 << 20));
  _Float16* Wkh = (_Float16*)(ws + ((size_t)26 << 20));
  _Float16* Wvh = (_Float16*)(ws + ((size_t)28 << 20));
  _Float16* Woh = (_Float16*)(ws + ((size_t)30 << 20));
  _Float16* Qp = (_Float16*)(ws + ((size_t)32 << 20)); // [B,H,S,D] 8MB
  _Float16* Kp = (_Float16*)(ws + ((size_t)40 << 20)); // [B,H,S,D] 8MB
  _Float16* Vp = (_Float16*)(ws + ((size_t)48 << 20)); // [B,H,D,S] 8MB (transposed)
  _Float16* ctx = Xq;  // queries-f16 no longer needed after Q projection

  const int nAct = Mrows * E;  // 4M
  const int nW   = E * E;      // 1M
  cvt_f32_to_f16<<<(nAct + 255) / 256, 256, 0, stream>>>(q, Xq, nAct);
  cvt_f32_to_f16<<<(nAct + 255) / 256, 256, 0, stream>>>(k, Xk, nAct);
  cvt_f32_to_f16<<<(nAct + 255) / 256, 256, 0, stream>>>(v, Xv, nAct);
  cvt_f32_to_f16<<<(nW + 255) / 256, 256, 0, stream>>>(Wq, Wqh, nW);
  cvt_f32_to_f16<<<(nW + 255) / 256, 256, 0, stream>>>(Wk, Wkh, nW);
  cvt_f32_to_f16<<<(nW + 255) / 256, 256, 0, stream>>>(Wv, Wvh, nW);
  cvt_f32_to_f16<<<(nW + 255) / 256, 256, 0, stream>>>(Wo, Woh, nW);

  dim3 ggrid(E / 128, Mrows / 128);  // (8, 32)
  gemm_nt_f16<1><<<ggrid, 256, 0, stream>>>(Xq, Wqh, Qp, Mrows, E, E);
  gemm_nt_f16<1><<<ggrid, 256, 0, stream>>>(Xk, Wkh, Kp, Mrows, E, E);
  gemm_nt_f16<2><<<ggrid, 256, 0, stream>>>(Xv, Wvh, Vp, Mrows, E, E);

  // Note: ctx aliases Xq; Q projection (reads Xq) completed before attn writes ctx.
  flash_attn_f16<<<Bb * H * (S / 128), 256, 0, stream>>>(Qp, Kp, Vp, ctx);

  gemm_nt_f16<0><<<ggrid, 256, 0, stream>>>(ctx, Woh, out, Mrows, E, E);
}
